// CausalSelfAttention_19533511262591
// MI455X (gfx1250) — matrix operand order repaired and checked by reading, not compiled, not run on hardware
//
#include <hip/hip_runtime.h>

// ---------------------------------------------------------------------------
// CDNA5 (gfx1250) causal self-attention, bf16 WMMA compute / f32 accumulate.
// ---------------------------------------------------------------------------

typedef __attribute__((ext_vector_type(16))) __bf16 v16bf;
typedef __attribute__((ext_vector_type(8)))  float  v8f;

#define DMODEL 1024
#define NHEADS 16
#define DHEAD  64
#define BATCH  2
#define SEQ    2048
#define MROWS  (BATCH * SEQ)   // 4096

__device__ __forceinline__ unsigned short f2bf(float f) {
  unsigned int u = __builtin_bit_cast(unsigned int, f);
  u += 0x7FFFu + ((u >> 16) & 1u);          // round-to-nearest-even
  return (unsigned short)(u >> 16);
}

union FragU {
  v16bf b;
  uint4 q[2];
};

// Build a 16-element bf16 fragment from two 16B chunks (global or LDS).
__device__ __forceinline__ v16bf load_frag(const unsigned short* p0,
                                           const unsigned short* p1) {
  FragU f;
  f.q[0] = *(const uint4*)p0;
  f.q[1] = *(const uint4*)p1;
  return f.b;
}

// All-ones bf16 fragment (1.0 = 0x3F80); used as B to row-sum P via WMMA.
__device__ __forceinline__ v16bf ones_frag() {
  FragU f;
  uint4 s;
  s.x = s.y = s.z = s.w = 0x3F803F80u;
  f.q[0] = s; f.q[1] = s;
  return f.b;
}

// ---------------------------------------------------------------------------
// Pass 0a: fp32 -> bf16 elementwise
// ---------------------------------------------------------------------------
__global__ void cvt_x_k(const float* __restrict__ x, unsigned short* __restrict__ xb) {
  int i = (blockIdx.x * 256 + threadIdx.x) * 4;
  float4 v = *(const float4*)(x + i);
  ushort4 o;
  o.x = f2bf(v.x); o.y = f2bf(v.y); o.z = f2bf(v.z); o.w = f2bf(v.w);
  *(ushort4*)(xb + i) = o;
}

// ---------------------------------------------------------------------------
// Pass 0b: W[k][n] fp32 -> Wt[n][k] bf16  (1024x1024)
// ---------------------------------------------------------------------------
__global__ void wT_cvt_k(const float* __restrict__ W, unsigned short* __restrict__ Wt) {
  int i = blockIdx.x * 256 + threadIdx.x;   // n fastest (coalesced read)
  int k = i >> 10;
  int n = i & 1023;
  Wt[(size_t)n * DMODEL + k] = f2bf(W[i]);
}

// ---------------------------------------------------------------------------
// Pass 1/3: C[4096x1024] = A[4096x1024](bf16) * Bt^T, Bt stored [N][K] bf16.
// Block tile 128x128, 8 waves (4 M x 2 N), wave tile 32x64, BK=32.
// MODE 0: bf16 scattered as [b][h][t][d]   (Q, K projections; oscale applied)
// MODE 1: bf16 scattered as [b][h][d][t]   (V^T projection)
// MODE 2: fp32 row-major [m][n]            (final output)
// ---------------------------------------------------------------------------
template <int MODE>
__global__ __launch_bounds__(256) void gemm_bf16_k(const unsigned short* __restrict__ A,
                                                   const unsigned short* __restrict__ Bt,
                                                   void* __restrict__ Cout,
                                                   float oscale) {
  __shared__ unsigned short As[128][40];   // +8 pad, rows 80B (16B-aligned chunks)
  __shared__ unsigned short Bs[128][40];

  const int tid  = threadIdx.x;
  const int w    = tid >> 5;
  const int lane = tid & 31;
  const int wm   = w >> 1;            // 0..3
  const int wn   = w & 1;             // 0..1
  const int hf   = lane >> 4;         // half-wave select
  const int l16  = lane & 15;
  const int m0   = blockIdx.y * 128;
  const int n0   = blockIdx.x * 128;

  v8f acc[2][4] = {};

  const int kq = tid & 3;             // which 8-elem chunk within BK=32
  const int r0 = tid >> 2;            // 0..63

  for (int kb = 0; kb < DMODEL; kb += 32) {
    // cooperative stage: 128x32 bf16 tiles of A and Bt
#pragma unroll
    for (int p = 0; p < 2; ++p) {
      int r = r0 + p * 64;
      *(uint4*)&As[r][kq * 8] =
          *(const uint4*)(A + (size_t)(m0 + r) * DMODEL + kb + kq * 8);
      *(uint4*)&Bs[r][kq * 8] =
          *(const uint4*)(Bt + (size_t)(n0 + r) * DMODEL + kb + kq * 8);
    }
    // prefetch next k-tile into caches while we compute this one
    if (kb + 32 < DMODEL) {
      __builtin_prefetch(A + (size_t)(m0 + r0) * DMODEL + kb + 32 + kq * 8, 0, 1);
      __builtin_prefetch(Bt + (size_t)(n0 + r0) * DMODEL + kb + 32 + kq * 8, 0, 1);
    }
    __syncthreads();

    // A fragment: lane=row(M), 16 contiguous K (lanes 0-15: K 0-15, 16-31: K 16-31)
    v16bf aF[2];
#pragma unroll
    for (int mt = 0; mt < 2; ++mt) {
      int row = wm * 32 + mt * 16 + l16;
      int o = hf * 16;
      aF[mt] = load_frag(&As[row][o], &As[row][o + 8]);
    }
    // B fragment: lane=col(N), 16 contiguous K (lanes 0-15: K 0-15, 16-31: K 16-31)
    v16bf bF[4];
#pragma unroll
    for (int nt = 0; nt < 4; ++nt) {
      int n = wn * 64 + nt * 16 + l16;
      int ko = hf * 16;
      bF[nt] = load_frag(&Bs[n][ko], &Bs[n][ko + 8]);
    }

#pragma unroll
    for (int mt = 0; mt < 2; ++mt)
#pragma unroll
      for (int nt = 0; nt < 4; ++nt)
        acc[mt][nt] = __builtin_amdgcn_wmma_f32_16x16x32_bf16(
            false, aF[mt], false, bF[nt], (short)0, acc[mt][nt], false, false);

    __syncthreads();
  }

  // epilogue (C layout: vgpr r -> row (hf? r+8 : r), lane%16 -> col)
#pragma unroll
  for (int mt = 0; mt < 2; ++mt)
#pragma unroll
    for (int nt = 0; nt < 4; ++nt)
#pragma unroll
      for (int r = 0; r < 8; ++r) {
        int grow = m0 + wm * 32 + mt * 16 + hf * 8 + r;
        int gcol = n0 + wn * 64 + nt * 16 + l16;
        float v = acc[mt][nt][r];
        if (MODE == 2) {
          ((float*)Cout)[(size_t)grow * DMODEL + gcol] = v;
        } else {
          int b = grow >> 11, t = grow & 2047;
          int h = gcol >> 6,  d = gcol & 63;
          size_t idx = (MODE == 0)
              ? ((((size_t)(b * NHEADS + h)) * SEQ + t) * DHEAD + d)
              : ((((size_t)(b * NHEADS + h)) * DHEAD + d) * SEQ + t);
          ((unsigned short*)Cout)[idx] = f2bf(v * oscale);
        }
      }
}

// ---------------------------------------------------------------------------
// Pass 2: per-wave flash attention (exp2 domain; scale pre-folded into Q).
// One wave per (b*h, 16-query tile). Keys processed in blocks of 32.
// ALL eight K-fragment chunks and four V fragments are loaded into distinct
// registers before the first QK WMMA so the scheduler can emit one load
// clause and stage the waits (no per-WMMA full-latency stalls).
// Q: [bh][t][64] bf16 (pre-scaled by log2e/sqrt(d)), K: [bh][t][64] bf16,
// Vt: [bh][64][t] bf16.  Y written bf16 row-major [b][t][h*64+d].
// ---------------------------------------------------------------------------
__global__ __launch_bounds__(32) void attn_k(const unsigned short* __restrict__ Q,
                                             const unsigned short* __restrict__ Kmat,
                                             const unsigned short* __restrict__ Vt,
                                             unsigned short* __restrict__ Yb) {
  __shared__ unsigned short Pls[16][40];   // P tile 16x32 bf16, padded rows

  const int lane = threadIdx.x;
  const int hf   = lane >> 4;
  const int l16  = lane & 15;
  const int bh   = blockIdx.y;            // b*16+h
  const int q0   = blockIdx.x * 16;

  const unsigned short* Qb = Q    + (size_t)bh * SEQ * DHEAD;
  const unsigned short* Kb = Kmat + (size_t)bh * SEQ * DHEAD;
  const unsigned short* Vb = Vt   + (size_t)bh * DHEAD * SEQ;

  // Resident Q A-fragments (K-dim 64 = two 32-wide steps)
  v16bf aQ[2];
  {
    int q = q0 + l16;
#pragma unroll
    for (int ks = 0; ks < 2; ++ks) {
      int o = ks * 32 + hf * 16;
      aQ[ks] = load_frag(Qb + (size_t)q * DHEAD + o,
                         Qb + (size_t)q * DHEAD + o + 8);
    }
  }
  const v16bf bOnes = ones_frag();

  v8f yacc[4] = {};
  float mrow[8], lrow[8];
#pragma unroll
  for (int r = 0; r < 8; ++r) { mrow[r] = -1e30f; lrow[r] = 0.f; }

  for (int kb = 0; kb < q0 + 16; kb += 32) {
    // ---- issue ALL global loads for this block up front, distinct regs
    v16bf kf[2][2];
#pragma unroll
    for (int sub = 0; sub < 2; ++sub) {
      int key = kb + sub * 16 + l16;
      const unsigned short* kp = Kb + (size_t)key * DHEAD + hf * 16;
      kf[sub][0] = load_frag(kp, kp + 8);            // d = ks0 half
      kf[sub][1] = load_frag(kp + 32, kp + 40);      // d = ks1 half
    }
    v16bf bV[4];
#pragma unroll
    for (int n = 0; n < 4; ++n) {
      const unsigned short* vp = Vb + (size_t)(n * 16 + l16) * SEQ + kb + hf * 16;
      bV[n] = load_frag(vp, vp + 8);
    }
    // prefetch next block's K rows and V row segments
    if (kb + 32 < q0 + 16) {
      __builtin_prefetch(Kb + (size_t)(kb + 32 + lane) * DHEAD, 0, 1);
      __builtin_prefetch(Vb + (size_t)lane * SEQ + kb + 32, 0, 1);
      __builtin_prefetch(Vb + (size_t)(lane + 32) * SEQ + kb + 32, 0, 1);
    }

    // ---- S = (Q K^T), already in exp2 domain (Q pre-scaled)
    v8f s[2];
#pragma unroll
    for (int sub = 0; sub < 2; ++sub) {
      v8f z = {};
      z = __builtin_amdgcn_wmma_f32_16x16x32_bf16(false, aQ[0], false, kf[sub][0],
                                                  (short)0, z, false, false);
      z = __builtin_amdgcn_wmma_f32_16x16x32_bf16(false, aQ[1], false, kf[sub][1],
                                                  (short)0, z, false, false);
      s[sub] = z;
    }

    // ---- causal mask only on diagonal blocks
    if (kb + 31 > q0) {
#pragma unroll
      for (int sub = 0; sub < 2; ++sub) {
        int key = kb + sub * 16 + l16;
#pragma unroll
        for (int r = 0; r < 8; ++r) {
          int qrow = q0 + hf * 8 + r;
          if (key > qrow) s[sub][r] = -1e30f;
        }
      }
    }

    // ---- row max (butterfly within 16-lane halves) + exp2
    float p0[8], p1[8], corr[8];
#pragma unroll
    for (int r = 0; r < 8; ++r) {
      float t = fmaxf(s[0][r], s[1][r]);
#pragma unroll
      for (int off = 1; off < 16; off <<= 1)
        t = fmaxf(t, __shfl_xor(t, off, 32));
      float mnew = fmaxf(mrow[r], t);
      corr[r] = exp2f(mrow[r] - mnew);
      p0[r] = exp2f(s[0][r] - mnew);
      p1[r] = exp2f(s[1][r] - mnew);
      mrow[r] = mnew;
#pragma unroll
      for (int n = 0; n < 4; ++n) yacc[n][r] *= corr[r];
    }

    // ---- C-layout -> A-fragment layout via LDS (in-wave handoff)
#pragma unroll
    for (int r = 0; r < 8; ++r) {
      Pls[hf * 8 + r][l16]      = f2bf(p0[r]);
      Pls[hf * 8 + r][l16 + 16] = f2bf(p1[r]);
    }
    asm volatile("s_wait_dscnt 0" ::: "memory");
    v16bf aP;
    {
      int o = hf * 16;
      aP = load_frag(&Pls[l16][o], &Pls[l16][o + 8]);
    }
    asm volatile("s_wait_dscnt 0" ::: "memory");

    // ---- row sum via WMMA against ones: every lane gets its rows' sums
    v8f rs = {};
    rs = __builtin_amdgcn_wmma_f32_16x16x32_bf16(false, aP, false, bOnes,
                                                 (short)0, rs, false, false);
#pragma unroll
    for (int r = 0; r < 8; ++r)
      lrow[r] = fmaf(lrow[r], corr[r], rs[r]);

    // ---- y += P * V (V fragments already resident)
#pragma unroll
    for (int n = 0; n < 4; ++n)
      yacc[n] = __builtin_amdgcn_wmma_f32_16x16x32_bf16(
          false, aP, false, bV[n], (short)0, yacc[n], false, false);
  }

  // ---- normalize and store y as bf16 [b][t][h*64+d]
  int b = bh >> 4, h = bh & 15;
#pragma unroll
  for (int r = 0; r < 8; ++r) {
    float inv = 1.0f / lrow[r];
    int qrow = q0 + hf * 8 + r;
    size_t base = ((size_t)(b * SEQ + qrow)) * DMODEL + h * DHEAD;
#pragma unroll
    for (int n = 0; n < 4; ++n)
      Yb[base + n * 16 + l16] = f2bf(yacc[n][r] * inv);
  }
}

// ---------------------------------------------------------------------------
// Launcher
// ---------------------------------------------------------------------------
extern "C" void kernel_launch(void* const* d_in, const int* in_sizes, int n_in,
                              void* d_out, int out_size, void* d_ws, size_t ws_size,
                              hipStream_t stream) {
  (void)in_sizes; (void)n_in; (void)out_size; (void)ws_size;

  const float* x  = (const float*)d_in[0];
  const float* Wq = (const float*)d_in[1];
  const float* Wk = (const float*)d_in[2];
  const float* Wv = (const float*)d_in[3];
  const float* Wo = (const float*)d_in[4];

  char* ws = (char*)d_ws;
  const size_t MB = 1024ull * 1024ull;
  unsigned short* xb  = (unsigned short*)(ws + 0);        // 8 MiB  [4096][1024]
  unsigned short* Wqt = (unsigned short*)(ws + 8 * MB);   // 2 MiB  [N][K]
  unsigned short* Wkt = (unsigned short*)(ws + 10 * MB);
  unsigned short* Wvt = (unsigned short*)(ws + 12 * MB);
  unsigned short* Wot = (unsigned short*)(ws + 14 * MB);
  unsigned short* Qb  = (unsigned short*)(ws + 16 * MB);  // 8 MiB [b][h][t][64]
  unsigned short* Kbf = (unsigned short*)(ws + 24 * MB);  // 8 MiB [b][h][t][64]
  unsigned short* Vtb = (unsigned short*)(ws + 32 * MB);  // 8 MiB [b][h][64][t]
  unsigned short* Yb  = (unsigned short*)(ws + 40 * MB);  // 8 MiB [4096][1024]

  // softmax scale folded into Q: (1/sqrt(64)) * log2(e)
  const float qScale = 0.125f * 1.44269504088896340736f;

  cvt_x_k<<<dim3(MROWS * DMODEL / (256 * 4)), dim3(256), 0, stream>>>(x, xb);
  wT_cvt_k<<<dim3(DMODEL * DMODEL / 256), dim3(256), 0, stream>>>(Wq, Wqt);
  wT_cvt_k<<<dim3(DMODEL * DMODEL / 256), dim3(256), 0, stream>>>(Wk, Wkt);
  wT_cvt_k<<<dim3(DMODEL * DMODEL / 256), dim3(256), 0, stream>>>(Wv, Wvt);
  wT_cvt_k<<<dim3(DMODEL * DMODEL / 256), dim3(256), 0, stream>>>(Wo, Wot);

  dim3 ggrid(DMODEL / 128, MROWS / 128);  // (8, 32)
  gemm_bf16_k<0><<<ggrid, dim3(256), 0, stream>>>(xb, Wqt, (void*)Qb, qScale);
  gemm_bf16_k<0><<<ggrid, dim3(256), 0, stream>>>(xb, Wkt, (void*)Kbf, 1.0f);
  gemm_bf16_k<1><<<ggrid, dim3(256), 0, stream>>>(xb, Wvt, (void*)Vtb, 1.0f);

  attn_k<<<dim3(SEQ / 16, BATCH * NHEADS), dim3(32), 0, stream>>>(Qb, Kbf, Vtb, Yb);

  gemm_bf16_k<2><<<ggrid, dim3(256), 0, stream>>>(Yb, Wot, d_out, 1.0f);
}
